// FasterRCNN_49667001811398
// MI455X (gfx1250) — compile-verified, matrix operand order checked by reading, not run on hardware
//
#include <hip/hip_runtime.h>
#include <hip/hip_bf16.h>

typedef __attribute__((ext_vector_type(16))) _Float16 v16h;
typedef __attribute__((ext_vector_type(8)))  _Float16 v8h;
typedef __attribute__((ext_vector_type(8)))  float    v8f;

// ---------------- constants (match reference) ----------------
static constexpr int   IMG        = 800;
static constexpr int   FM         = 50;
static constexpr int   STRIDE     = 16;
static constexpr int   A_NUM      = 9;
static constexpr int   NUM_CLASSES= 21;
static constexpr int   PRE_NMS_K  = 2000;
static constexpr int   POST_NMS_K = 300;
static constexpr float NMS_THRESH = 0.7f;
static constexpr float MIN_SIZE   = 16.0f;
static constexpr int   POOLP      = 7;
static constexpr float NEGC       = -1e30f;
static constexpr int   NANCH      = FM * FM * A_NUM;     // 22500
static constexpr int   FCK        = POOLP * POOLP * 512; // 25088
static constexpr int   BST        = 40;                  // LDS B row stride (halves), bank-spread pad

// ---------------- weight convert f32 -> f16 with transpose to B-major ----------------
// src: [taps, srcCin, Cout] (HWIO, n contiguous)  ->  dst: [taps, Cout, Cin] (k contiguous)
// Zero-pads srcCin -> Cin. Used for all conv and FC weights (FC: taps=1, Cin=K, Cout=N).
__global__ void cvt_transpose_w(const float* __restrict__ src, _Float16* __restrict__ dst,
                                int taps, int Cin, int Cout, int srcCin) {
    size_t total = (size_t)taps * Cout * Cin;
    size_t i = (size_t)blockIdx.x * blockDim.x + threadIdx.x;
    size_t stride = (size_t)gridDim.x * blockDim.x;
    for (; i < total; i += stride) {
        int c   = (int)(i % Cin);
        int n   = (int)((i / Cin) % Cout);
        int tap = (int)(i / ((size_t)Cin * Cout));
        dst[i] = (c < srcCin) ? (_Float16)src[((size_t)tap * srcCin + c) * Cout + n]
                              : (_Float16)0.f;
    }
}

// pad image HWC 3 -> 32 channels, f16
__global__ void pad_image_f16(const float* __restrict__ img, _Float16* __restrict__ out, int HW) {
    int i = blockIdx.x * blockDim.x + threadIdx.x;
    int stride = gridDim.x * blockDim.x;
    for (; i < HW * 32; i += stride) {
        int c = i & 31, p = i >> 5;
        out[i] = (c < 3) ? (_Float16)img[p * 3 + c] : (_Float16)0.f;
    }
}

__device__ __forceinline__ v16h cat16(v8h lo, v8h hi) {
    return __builtin_shufflevector(lo, hi, 0, 1, 2, 3, 4, 5, 6, 7, 8, 9, 10, 11, 12, 13, 14, 15);
}

// ---------------- 3x3 conv, implicit GEMM via WMMA (requires Cin % 32 == 0) ----------------
// block = 256 (8 waves). wave wv -> 16-pixel tile, all waves share a 64-wide Cout group.
// Weights pre-transposed [tap][n][c]; K-chunk staged to LDS Bs[n][k] with one
// v8h global load + one ds_store_b128 per thread. A loads are branch-free:
// clamped-safe address + cndmask select against zero (keeps acc regs pinned).
__global__ void conv3x3_wmma(const _Float16* __restrict__ in, const _Float16* __restrict__ w,
                             const float* __restrict__ bias, _Float16* __restrict__ out,
                             int H, int W, int Cin, int Cout, int relu) {
    __shared__ _Float16 Bs[64 * BST];
    const int lane  = threadIdx.x & 31;
    const int wv    = threadIdx.x >> 5;
    const int mbase = (blockIdx.x * 8 + wv) * 16;
    const int n0    = blockIdx.y * 64;
    const int HW    = H * W;
    const int row   = lane & 15;
    const int kb    = (lane >> 4) * 8;
    const int m     = mbase + row;
    const int msafe = min(m, HW - 1);
    const int ybase = msafe / W, xbase = msafe % W;
    // staging role: 4 consecutive threads cover one 64B row of the chunk
    const int sn = threadIdx.x >> 2;   // 0..63  (n within group)
    const int sj = threadIdx.x & 3;    // 0..3   (16B segment within k-chunk)
    const bool wok = (n0 + sn) < Cout;
    const _Float16* wstage = w + ((size_t)(n0 + (wok ? sn : 0))) * Cin + sj * 8;

    v8f acc[4] = {{}, {}, {}, {}};

    for (int ky = 0; ky < 3; ++ky) {
        for (int kx = 0; kx < 3; ++kx) {
            const int tap = ky * 3 + kx;
            const int y = ybase + ky - 1;
            const int x = xbase + kx - 1;
            const bool inb = (m < HW) && (y >= 0) && (y < H) && (x >= 0) && (x < W);
            const int ysafe = min(max(y, 0), H - 1);
            const int xsafe = min(max(x, 0), W - 1);
            const _Float16* srcbase = in + ((size_t)ysafe * W + xsafe) * Cin + kb;
            const _Float16* wtap = wstage + (size_t)tap * Cout * Cin;
            for (int c0 = 0; c0 < Cin; c0 += 32) {
                // stage B chunk: Bs[n][k] <- wT[tap][n0+n][c0+k], vectorized 16B/16B
                v8h wz = {};
                v8h wvv = *(const v8h*)(wtap + c0);
                *(v8h*)&Bs[sn * BST + sj * 8] = wok ? wvv : wz;
                __syncthreads();

                // branch-free predicated A fragment
                const _Float16* src = srcbase + c0;
                v16h az = {};
                v16h al = cat16(*(const v8h*)src, *(const v8h*)(src + 16));
                v16h a  = inb ? al : az;
#pragma unroll
                for (int s = 0; s < 4; ++s) {
                    const _Float16* bp = &Bs[(s * 16 + row) * BST + kb];
                    v16h b = cat16(*(const v8h*)bp, *(const v8h*)(bp + 16));
                    acc[s] = __builtin_amdgcn_wmma_f32_16x16x32_f16(false, a, false, b,
                                                                    (short)0, acc[s], false, false);
                }
                __syncthreads();
            }
        }
    }

    // C/D layout: VGPR r -> M = r (lanes 0-15) or r+8 (lanes 16-31), N = lane%16
#pragma unroll
    for (int s = 0; s < 4; ++s) {
        const int n = n0 + s * 16 + row;
        if (n >= Cout) continue;
#pragma unroll
        for (int r = 0; r < 8; ++r) {
            int m2 = mbase + r + ((lane >> 4) << 3);
            if (m2 < HW) {
                float v = acc[s][r] + bias[n];
                if (relu) v = fmaxf(v, 0.f);
                out[(size_t)m2 * Cout + n] = (_Float16)v;
            }
        }
    }
}

// ---------------- 2x2 max-pool stride 2 ----------------
__global__ void maxpool2_f16(const _Float16* __restrict__ in, _Float16* __restrict__ out,
                             int H, int W, int C) {
    int Ho = H >> 1, Wo = W >> 1;
    size_t total = (size_t)Ho * Wo * C;
    size_t i = (size_t)blockIdx.x * blockDim.x + threadIdx.x;
    size_t stride = (size_t)gridDim.x * blockDim.x;
    for (; i < total; i += stride) {
        int c = (int)(i % C);
        int x = (int)((i / C) % Wo);
        int y = (int)(i / ((size_t)C * Wo));
        const _Float16* p = in + ((size_t)(2 * y) * W + 2 * x) * C + c;
        float v0 = (float)p[0];
        float v1 = (float)p[C];
        float v2 = (float)p[(size_t)W * C];
        float v3 = (float)p[(size_t)W * C + C];
        out[i] = (_Float16)fmaxf(fmaxf(v0, v1), fmaxf(v2, v3));
    }
}

// ---------------- 1x1 conv (RPN heads, tiny) ----------------
__global__ void conv1x1_f32(const _Float16* __restrict__ in, const float* __restrict__ w,
                            const float* __restrict__ b, float* __restrict__ out,
                            int HW, int Cin, int Cout) {
    int i = blockIdx.x * blockDim.x + threadIdx.x;
    int stride = gridDim.x * blockDim.x;
    for (; i < HW * Cout; i += stride) {
        int co = i % Cout, p = i / Cout;
        const _Float16* src = in + (size_t)p * Cin;
        float s = b[co];
        for (int k = 0; k < Cin; ++k) s += (float)src[k] * w[(size_t)k * Cout + co];
        out[i] = s;
    }
}

// ---------------- RPN: softmax score + anchor decode + min-size filter ----------------
__global__ void rpn_postprocess(const float* __restrict__ cls, const float* __restrict__ box,
                                float* __restrict__ scores, float* __restrict__ boxes) {
    int i = blockIdx.x * blockDim.x + threadIdx.x;
    if (i >= NANCH) return;
    int cell = i / A_NUM, a = i % A_NUM;
    int r = a / 3, s = a % 3;
    const float scales[3] = {128.f, 256.f, 512.f};
    const float ratios[3] = {0.5f, 1.f, 2.f};
    float aw = scales[s] * sqrtf(1.0f / ratios[r]);
    float ah = scales[s] * sqrtf(ratios[r]);
    float acy = ((float)(cell / FM) + 0.5f) * (float)STRIDE;
    float acx = ((float)(cell % FM) + 0.5f) * (float)STRIDE;

    float l0 = cls[(size_t)cell * (A_NUM * 2) + a * 2 + 0];
    float l1 = cls[(size_t)cell * (A_NUM * 2) + a * 2 + 1];
    float score = 1.0f / (1.0f + expf(l0 - l1));

    const float* d = box + (size_t)cell * (A_NUM * 4) + a * 4;
    float cx = acx + d[0] * aw;
    float cy = acy + d[1] * ah;
    float w  = aw * expf(fminf(fmaxf(d[2], -6.f), 6.f));
    float h  = ah * expf(fminf(fmaxf(d[3], -6.f), 6.f));
    float x1 = fminf(fmaxf(cx - 0.5f * w, 0.f), (float)IMG);
    float y1 = fminf(fmaxf(cy - 0.5f * h, 0.f), (float)IMG);
    float x2 = fminf(fmaxf(cx + 0.5f * w, 0.f), (float)IMG);
    float y2 = fminf(fmaxf(cy + 0.5f * h, 0.f), (float)IMG);
    bool keep = ((x2 - x1) >= MIN_SIZE) && ((y2 - y1) >= MIN_SIZE);
    scores[i] = keep ? score : NEGC;
    boxes[i * 4 + 0] = x1; boxes[i * 4 + 1] = y1;
    boxes[i * 4 + 2] = x2; boxes[i * 4 + 3] = y2;
}

// ---------------- top-k by iterative parallel argmax (single block) ----------------
__global__ void topk_kernel(float* __restrict__ scores, const float* __restrict__ boxes,
                            float* __restrict__ tb, float* __restrict__ ts, int N, int K) {
    __shared__ float ssc[1024];
    __shared__ int   sid[1024];
    int tid = threadIdx.x;
    for (int it = 0; it < K; ++it) {
        float best = NEGC * 2.f; int bi = 0x7fffffff;
        for (int i = tid; i < N; i += blockDim.x) {
            float s = scores[i];
            if (s > best || (s == best && i < bi)) { best = s; bi = i; }
        }
        ssc[tid] = best; sid[tid] = bi;
        __syncthreads();
        for (int off = blockDim.x >> 1; off > 0; off >>= 1) {
            if (tid < off) {
                if (ssc[tid + off] > ssc[tid] ||
                    (ssc[tid + off] == ssc[tid] && sid[tid + off] < sid[tid])) {
                    ssc[tid] = ssc[tid + off]; sid[tid] = sid[tid + off];
                }
            }
            __syncthreads();
        }
        if (tid == 0) {
            int idx = sid[0];
            ts[it] = ssc[0];
            tb[it * 4 + 0] = boxes[idx * 4 + 0];
            tb[it * 4 + 1] = boxes[idx * 4 + 1];
            tb[it * 4 + 2] = boxes[idx * 4 + 2];
            tb[it * 4 + 3] = boxes[idx * 4 + 3];
            scores[idx] = NEGC * 2.f;
        }
        __syncthreads();
    }
}

// ---------------- NMS (single block, 300 sequential selections) ----------------
__global__ void nms_kernel(const float* __restrict__ boxes, float* __restrict__ sc,
                           float* __restrict__ ob, float* __restrict__ os) {
    __shared__ float ssc[256];
    __shared__ int   sid[256];
    __shared__ float bb[4];
    int tid = threadIdx.x;
    for (int it = 0; it < POST_NMS_K; ++it) {
        float best = NEGC * 2.f; int bi = 0x7fffffff;
        for (int i = tid; i < PRE_NMS_K; i += blockDim.x) {
            float s = sc[i];
            if (s > best || (s == best && i < bi)) { best = s; bi = i; }
        }
        ssc[tid] = best; sid[tid] = bi;
        __syncthreads();
        for (int off = 128; off > 0; off >>= 1) {
            if (tid < off) {
                if (ssc[tid + off] > ssc[tid] ||
                    (ssc[tid + off] == ssc[tid] && sid[tid + off] < sid[tid])) {
                    ssc[tid] = ssc[tid + off]; sid[tid] = sid[tid + off];
                }
            }
            __syncthreads();
        }
        if (tid == 0) {
            int idx = sid[0];
            float s = ssc[0];
            bool valid = s > NEGC * 0.5f;
            bb[0] = boxes[idx * 4 + 0]; bb[1] = boxes[idx * 4 + 1];
            bb[2] = boxes[idx * 4 + 2]; bb[3] = boxes[idx * 4 + 3];
            ob[it * 4 + 0] = valid ? bb[0] : 0.f;
            ob[it * 4 + 1] = valid ? bb[1] : 0.f;
            ob[it * 4 + 2] = valid ? bb[2] : 0.f;
            ob[it * 4 + 3] = valid ? bb[3] : 0.f;
            os[it] = valid ? s : 0.f;
            sc[idx] = NEGC;
        }
        __syncthreads();
        float a1 = fmaxf(bb[2] - bb[0], 0.f) * fmaxf(bb[3] - bb[1], 0.f);
        for (int i = tid; i < PRE_NMS_K; i += blockDim.x) {
            float xx1 = fmaxf(bb[0], boxes[i * 4 + 0]);
            float yy1 = fmaxf(bb[1], boxes[i * 4 + 1]);
            float xx2 = fminf(bb[2], boxes[i * 4 + 2]);
            float yy2 = fminf(bb[3], boxes[i * 4 + 3]);
            float inter = fmaxf(xx2 - xx1, 0.f) * fmaxf(yy2 - yy1, 0.f);
            float a2 = fmaxf(boxes[i * 4 + 2] - boxes[i * 4 + 0], 0.f) *
                       fmaxf(boxes[i * 4 + 3] - boxes[i * 4 + 1], 0.f);
            float iou = inter / (a1 + a2 - inter + 1e-8f);
            if (iou >= NMS_THRESH) sc[i] = NEGC;
        }
        __syncthreads();
    }
}

// ---------------- ROI-align into packed f16 GEMM-A matrix [300, 25088] ----------------
__global__ void roi_align_kernel(const _Float16* __restrict__ fm, const float* __restrict__ boxes,
                                 _Float16* __restrict__ Amat) {
    int cell = blockIdx.x;                 // n*49 + py*7 + px
    int n  = cell / (POOLP * POOLP);
    int pp = cell % (POOLP * POOLP);
    int py = pp / POOLP, px = pp % POOLP;
    float x1 = boxes[n * 4 + 0] / STRIDE, y1 = boxes[n * 4 + 1] / STRIDE;
    float x2 = boxes[n * 4 + 2] / STRIDE, y2 = boxes[n * 4 + 3] / STRIDE;
    float gy = y1 + ((py + 0.5f) / POOLP) * (y2 - y1) - 0.5f;
    float gx = x1 + ((px + 0.5f) / POOLP) * (x2 - x1) - 0.5f;
    float fy = floorf(gy), fx = floorf(gx);
    float wy = gy - fy, wx = gx - fx;
    int y0 = min(max((int)fy, 0), FM - 1);
    int y1i = min(max((int)fy + 1, 0), FM - 1);
    int x0 = min(max((int)fx, 0), FM - 1);
    int x1i = min(max((int)fx + 1, 0), FM - 1);
    const _Float16* p00 = fm + ((size_t)y0 * FM + x0) * 512;
    const _Float16* p01 = fm + ((size_t)y0 * FM + x1i) * 512;
    const _Float16* p10 = fm + ((size_t)y1i * FM + x0) * 512;
    const _Float16* p11 = fm + ((size_t)y1i * FM + x1i) * 512;
    _Float16* dst = Amat + (size_t)cell * 512;
    for (int c = threadIdx.x; c < 512; c += blockDim.x) {
        float v = (float)p00[c] * (1.f - wy) * (1.f - wx) +
                  (float)p01[c] * (1.f - wy) * wx +
                  (float)p10[c] * wy * (1.f - wx) +
                  (float)p11[c] * wy * wx;
        dst[c] = (_Float16)v;
    }
}

// ---------------- generic GEMM (f16 x f16 -> f32) via WMMA ----------------
// A [M,K] f16 (K % 32 == 0), BT [N,K] f16 (pre-transposed), out f16 or f32.
// Wave tile 16x64 (4 accumulators on one A fragment), vectorized LDS staging,
// branch-free predicated A loads.
__global__ void gemm_wmma(const _Float16* __restrict__ A, const _Float16* __restrict__ BT,
                          const float* __restrict__ bias, _Float16* __restrict__ outh,
                          float* __restrict__ outf, int M, int N, int K, int Nld,
                          int relu, int out_f32) {
    __shared__ _Float16 Bs[64 * BST];
    const int lane  = threadIdx.x & 31;
    const int wv    = threadIdx.x >> 5;
    const int mbase = (blockIdx.x * 8 + wv) * 16;
    const int n0    = blockIdx.y * 64;
    const int row   = lane & 15;
    const int kb    = (lane >> 4) * 8;
    const int m     = mbase + row;
    const bool inb  = (m < M);
    const _Float16* arow = A + (size_t)min(m, M - 1) * K + kb;
    const int sn = threadIdx.x >> 2;   // 0..63
    const int sj = threadIdx.x & 3;    // 0..3
    const bool wok = (n0 + sn) < N;
    const _Float16* wrow = BT + (size_t)(n0 + (wok ? sn : 0)) * K + sj * 8;

    v8f acc[4] = {{}, {}, {}, {}};
    for (int kc = 0; kc < K; kc += 32) {
        if (threadIdx.x == 0 && (kc + 32) < K)
            __builtin_prefetch(wrow + kc + 32, 0, 1);   // global_prefetch_b8
        v8h wz = {};
        v8h wvv = *(const v8h*)(wrow + kc);
        *(v8h*)&Bs[sn * BST + sj * 8] = wok ? wvv : wz;
        __syncthreads();
        const _Float16* src = arow + kc;
        v16h az = {};
        v16h al = cat16(*(const v8h*)src, *(const v8h*)(src + 16));
        v16h a  = inb ? al : az;
#pragma unroll
        for (int s = 0; s < 4; ++s) {
            const _Float16* bp = &Bs[(s * 16 + row) * BST + kb];
            v16h b = cat16(*(const v8h*)bp, *(const v8h*)(bp + 16));
            acc[s] = __builtin_amdgcn_wmma_f32_16x16x32_f16(false, a, false, b,
                                                            (short)0, acc[s], false, false);
        }
        __syncthreads();
    }
#pragma unroll
    for (int s = 0; s < 4; ++s) {
        const int n = n0 + s * 16 + row;
        if (n >= N) continue;
#pragma unroll
        for (int r = 0; r < 8; ++r) {
            int m2 = mbase + r + ((lane >> 4) << 3);
            if (m2 < M) {
                float v = acc[s][r] + bias[n];
                if (relu) v = fmaxf(v, 0.f);
                if (out_f32) outf[(size_t)m2 * Nld + n] = v;
                else         outh[(size_t)m2 * Nld + n] = (_Float16)v;
            }
        }
    }
}

// ================================================================ host side
static const int vggCi[13] = {32,64,64,128,128,256,256,256,512,512,512,512,512}; // layer0 padded 3->32
static const int vggCo[13] = {64,64,128,128,256,256,256,512,512,512,512,512,512};

extern "C" void kernel_launch(void* const* d_in, const int* in_sizes, int n_in,
                              void* d_out, int out_size, void* d_ws, size_t ws_size,
                              hipStream_t stream) {
    (void)in_sizes; (void)n_in; (void)out_size; (void)ws_size;
    // ---- inputs (setup_inputs dict order) ----
    const float* images = (const float*)d_in[0];
    const float* vggW[13]; const float* vggB[13];
    for (int i = 0; i < 13; ++i) { vggW[i] = (const float*)d_in[1 + i]; vggB[i] = (const float*)d_in[14 + i]; }
    const float* rpnConvW = (const float*)d_in[27];
    const float* rpnConvB = (const float*)d_in[28];
    const float* rpnClsW  = (const float*)d_in[29];
    const float* rpnClsB  = (const float*)d_in[30];
    const float* rpnBoxW  = (const float*)d_in[31];
    const float* rpnBoxB  = (const float*)d_in[32];
    const float* fc1W = (const float*)d_in[33]; const float* fc1B = (const float*)d_in[34];
    const float* fc2W = (const float*)d_in[35]; const float* fc2B = (const float*)d_in[36];
    const float* clsW = (const float*)d_in[37]; const float* clsB = (const float*)d_in[38];
    const float* boxW = (const float*)d_in[39]; const float* boxB = (const float*)d_in[40];
    float* out = (float*)d_out;

    // ---- workspace carve (deterministic) ----
    char* ws = (char*)d_ws;
    size_t cur = 0;
    auto alloc = [&](size_t bytes) -> char* {
        char* p = ws + cur;
        cur += (bytes + 255) & ~(size_t)255;
        return p;
    };
    _Float16* w16[13];
    for (int i = 0; i < 13; ++i)
        w16[i] = (_Float16*)alloc((size_t)9 * vggCi[i] * vggCo[i] * sizeof(_Float16));
    _Float16* wRpn = (_Float16*)alloc((size_t)9 * 512 * 512 * sizeof(_Float16));
    _Float16* wFc1 = (_Float16*)alloc((size_t)FCK * 1024 * sizeof(_Float16));
    _Float16* wFc2 = (_Float16*)alloc((size_t)1024 * 1024 * sizeof(_Float16));
    _Float16* wCls = (_Float16*)alloc((size_t)1024 * NUM_CLASSES * sizeof(_Float16));
    _Float16* wBox = (_Float16*)alloc((size_t)1024 * NUM_CLASSES * 4 * sizeof(_Float16));
    size_t actElems = (size_t)IMG * IMG * 64;                 // max activation footprint
    _Float16* actA = (_Float16*)alloc(actElems * sizeof(_Float16));
    _Float16* actB = (_Float16*)alloc(actElems * sizeof(_Float16));
    _Float16* rpnc = (_Float16*)alloc((size_t)FM * FM * 512 * sizeof(_Float16));
    float* clsOut  = (float*)alloc((size_t)FM * FM * A_NUM * 2 * sizeof(float));
    float* boxOut  = (float*)alloc((size_t)FM * FM * A_NUM * 4 * sizeof(float));
    float* scores  = (float*)alloc((size_t)NANCH * sizeof(float));
    float* pboxes  = (float*)alloc((size_t)NANCH * 4 * sizeof(float));
    float* tkBoxes = (float*)alloc((size_t)PRE_NMS_K * 4 * sizeof(float));
    float* tkScores= (float*)alloc((size_t)PRE_NMS_K * sizeof(float));
    float* nmsBoxes= (float*)alloc((size_t)POST_NMS_K * 4 * sizeof(float));
    float* nmsScores=(float*)alloc((size_t)POST_NMS_K * sizeof(float));
    _Float16* Amat = (_Float16*)alloc((size_t)POST_NMS_K * FCK * sizeof(_Float16));
    _Float16* h1   = (_Float16*)alloc((size_t)POST_NMS_K * 1024 * sizeof(_Float16));
    _Float16* h2   = (_Float16*)alloc((size_t)POST_NMS_K * 1024 * sizeof(_Float16));

    // ---- weight conversion: f32 -> f16, transposed to [tap][n][k] (layer0 padded) ----
    cvt_transpose_w<<<72, 256, 0, stream>>>(vggW[0], w16[0], 9, 32, 64, 3);
    for (int i = 1; i < 13; ++i)
        cvt_transpose_w<<<1024, 256, 0, stream>>>(vggW[i], w16[i], 9, vggCi[i], vggCo[i], vggCi[i]);
    cvt_transpose_w<<<2048, 256, 0, stream>>>(rpnConvW, wRpn, 9, 512, 512, 512);
    cvt_transpose_w<<<4096, 256, 0, stream>>>(fc1W, wFc1, 1, FCK, 1024, FCK);
    cvt_transpose_w<<<1024, 256, 0, stream>>>(fc2W, wFc2, 1, 1024, 1024, 1024);
    cvt_transpose_w<<<128, 256, 0, stream>>>(clsW, wCls, 1, 1024, NUM_CLASSES, 1024);
    cvt_transpose_w<<<256, 256, 0, stream>>>(boxW, wBox, 1, 1024, NUM_CLASSES * 4, 1024);
    pad_image_f16<<<4096, 256, 0, stream>>>(images, actA, IMG * IMG);

    // ---- VGG backbone: 13 conv layers, pool after layers 1,3,6,9 ----
    int H = IMG, W = IMG;
    _Float16* pin = actA; _Float16* pout = actB;
    for (int i = 0; i < 13; ++i) {
        int HW = H * W;
        int pixTiles = (HW + 15) / 16;
        dim3 grid((pixTiles + 7) / 8, vggCo[i] / 64);
        conv3x3_wmma<<<grid, 256, 0, stream>>>(pin, w16[i], vggB[i], pout,
                                               H, W, vggCi[i], vggCo[i], 1);
        _Float16* t = pin; pin = pout; pout = t;
        if (i == 1 || i == 3 || i == 6 || i == 9) {
            maxpool2_f16<<<2048, 256, 0, stream>>>(pin, pout, H, W, vggCo[i]);
            t = pin; pin = pout; pout = t;
            H >>= 1; W >>= 1;
        }
    }
    _Float16* fm = pin;  // [50*50, 512] f16

    // ---- RPN ----
    {
        int pixTiles = (FM * FM + 15) / 16;
        dim3 grid((pixTiles + 7) / 8, 512 / 64);
        conv3x3_wmma<<<grid, 256, 0, stream>>>(fm, wRpn, rpnConvB, rpnc, FM, FM, 512, 512, 1);
    }
    conv1x1_f32<<<(FM * FM * 18 + 255) / 256, 256, 0, stream>>>(rpnc, rpnClsW, rpnClsB, clsOut, FM * FM, 512, 18);
    conv1x1_f32<<<(FM * FM * 36 + 255) / 256, 256, 0, stream>>>(rpnc, rpnBoxW, rpnBoxB, boxOut, FM * FM, 512, 36);
    rpn_postprocess<<<(NANCH + 255) / 256, 256, 0, stream>>>(clsOut, boxOut, scores, pboxes);

    // ---- top-k + NMS ----
    topk_kernel<<<1, 1024, 0, stream>>>(scores, pboxes, tkBoxes, tkScores, NANCH, PRE_NMS_K);
    nms_kernel<<<1, 256, 0, stream>>>(tkBoxes, tkScores, nmsBoxes, nmsScores);

    // ---- ROI align into GEMM-A matrix ----
    roi_align_kernel<<<POST_NMS_K * POOLP * POOLP, 256, 0, stream>>>(fm, nmsBoxes, Amat);

    // ---- FC head via WMMA GEMMs ----
    int mTiles = (POST_NMS_K + 15) / 16;            // 19
    int gx = (mTiles + 7) / 8;                      // 3
    gemm_wmma<<<dim3(gx, 1024 / 64), 256, 0, stream>>>(Amat, wFc1, fc1B, h1, nullptr,
                                                       POST_NMS_K, 1024, FCK, 1024, 1, 0);
    gemm_wmma<<<dim3(gx, 1024 / 64), 256, 0, stream>>>(h1, wFc2, fc2B, h2, nullptr,
                                                       POST_NMS_K, 1024, 1024, 1024, 1, 0);
    // logits -> d_out[0 : 300*21]
    gemm_wmma<<<dim3(gx, 1), 256, 0, stream>>>(
        h2, wCls, clsB, nullptr, out, POST_NMS_K, NUM_CLASSES, 1024, NUM_CLASSES, 0, 1);
    // box deltas -> d_out[300*21 : 300*21 + 300*84]
    gemm_wmma<<<dim3(gx, (NUM_CLASSES * 4 + 63) / 64), 256, 0, stream>>>(
        h2, wBox, boxB, nullptr, out + POST_NMS_K * NUM_CLASSES,
        POST_NMS_K, NUM_CLASSES * 4, 1024, NUM_CLASSES * 4, 0, 1);
}
